// CustomBSplineLayer_26182120636498
// MI455X (gfx1250) — matrix operand order repaired and checked by reading, not compiled, not run on hardware
//
#include <hip/hip_runtime.h>
#include <hip/hip_bf16.h>

// ---------------------------------------------------------------------------
// KAN B-spline layer for MI455X (gfx1250):
//   out[b,o] = sum_{i,g} basis(x[b,i])[g] * coef[o,i,g] * w[o,i]
// restructured as  S[8192 x 4096] (f16)  @  W2^T[4096 x 512] (f16) -> f32.
// GEMM: v_wmma_f32_16x16x32_f16, triple-buffered LDS tiles fed by
// global_load_async_to_lds_b128 (ASYNCcnt), one split-barrier per K-step.
// ---------------------------------------------------------------------------

typedef _Float16 v16h __attribute__((ext_vector_type(16)));
typedef _Float16 v8h  __attribute__((ext_vector_type(8)));
typedef float    v8f  __attribute__((ext_vector_type(8)));
typedef int      i4v  __attribute__((vector_size(16)));   // int4, matches builtin

#define AS1 __attribute__((address_space(1)))
#define AS3 __attribute__((address_space(3)))

#define BATCH   8192
#define INSZ    512
#define OUTSZ   512
#define GRIDC   8
#define KDIM    (INSZ * GRIDC)   // 4096

#define BM 128
#define BN 128
#define BK 64
#define LDAH 72                  // padded LDS row stride in halves (144 B = 9*16 B)
#define NSTG 3                   // triple buffer

// ---------------------------------------------------------------------------
// Phase 1: W2[o, i*8+g] = coef[o,i,g] * w[o,i]   (f32 -> f16)
// ---------------------------------------------------------------------------
__global__ __launch_bounds__(256) void kan_fuse_w(const float* __restrict__ w,
                                                  const float* __restrict__ coef,
                                                  _Float16* __restrict__ W2) {
  const int idx = blockIdx.x * 256 + threadIdx.x;      // o*INSZ + i
  const float ww = w[idx];
  const float4* c4 = (const float4*)(coef + (size_t)idx * GRIDC);
  const float4 lo = c4[0];
  const float4 hi = c4[1];
  union { _Float16 h[8]; uint4 u; } p;
  p.h[0] = (_Float16)(lo.x * ww);
  p.h[1] = (_Float16)(lo.y * ww);
  p.h[2] = (_Float16)(lo.z * ww);
  p.h[3] = (_Float16)(lo.w * ww);
  p.h[4] = (_Float16)(hi.x * ww);
  p.h[5] = (_Float16)(hi.y * ww);
  p.h[6] = (_Float16)(hi.z * ww);
  p.h[7] = (_Float16)(hi.w * ww);
  *(uint4*)(W2 + (size_t)idx * GRIDC) = p.u;
}

// ---------------------------------------------------------------------------
// Phase 2: S[b, i*8+g] = cubic B-spline basis of clamp(x[b,i]) via De Boor.
// Uniform knots: c_j = -1 + j*(2/7), j = 0..11.
// ---------------------------------------------------------------------------
__global__ __launch_bounds__(256) void kan_basis(const float* __restrict__ x,
                                                 _Float16* __restrict__ S) {
  const size_t idx = (size_t)blockIdx.x * 256 + threadIdx.x;  // b*INSZ + i
  float xv = x[idx];
  xv = fminf(fmaxf(xv, -1.0f), 1.0f);

  const float h = 2.0f / 7.0f;
  float b[11];
#pragma unroll
  for (int j = 0; j < 11; ++j) {
    const float cj  = -1.0f + (float)j * h;
    const float cj1 = -1.0f + (float)(j + 1) * h;
    b[j] = (xv >= cj && xv < cj1) ? 1.0f : 0.0f;
  }
#pragma unroll
  for (int k = 1; k <= 3; ++k) {
    const float inv = 1.0f / ((float)k * h);
#pragma unroll 10
    for (int j = 0; j < 11 - k; ++j) {
      const float cj   = -1.0f + (float)j * h;
      const float cjk1 = -1.0f + (float)(j + k + 1) * h;
      b[j] = (xv - cj) * inv * b[j] + (cjk1 - xv) * inv * b[j + 1];
    }
  }

  union { _Float16 h8[8]; uint4 u; } p;
#pragma unroll
  for (int g = 0; g < 8; ++g) p.h8[g] = (_Float16)b[g];
  *(uint4*)(S + idx * GRIDC) = p.u;
}

// ---------------------------------------------------------------------------
// Phase 3: WMMA GEMM  out[8192 x 512] = S[8192 x 4096] * W2^T
// 256 threads = 8 waves; tile 128x128x64; wave tile 32x64 (2x4 accumulators).
// Triple-buffered async producer: stages k, k+1, k+2 in flight.
// ---------------------------------------------------------------------------
union F16x16 { v16h v; v8h h[2]; };

__global__ __launch_bounds__(256) void kan_wmma_gemm(const _Float16* __restrict__ S,
                                                     const _Float16* __restrict__ W2,
                                                     float* __restrict__ out) {
  __shared__ alignas(16) _Float16 AsBuf[NSTG][BM * LDAH];
  __shared__ alignas(16) _Float16 BsBuf[NSTG][BN * LDAH];

  const int t    = threadIdx.x;
  const int lane = t & 31;
  const int wv   = t >> 5;          // wave 0..7
  const int wm   = wv & 3;          // 4 M-sub-tiles of 32 rows
  const int wn   = wv >> 2;         // 2 N-sub-tiles of 64 cols
  const int half = lane >> 4;       // 0 / 1
  const int lr   = lane & 15;

  const int mBase = blockIdx.y * BM;
  const int nBase = blockIdx.x * BN;

  // Staging chunks: tile = 128 rows x 64 halves = 1024 x 16B; 4 chunks/thread.
  int srow[4], skp[4];
#pragma unroll
  for (int c = 0; c < 4; ++c) {
    const int ch = t + c * 256;
    srow[c] = ch >> 3;
    skp[c]  = (ch & 7) << 3;
  }

  // Issue one stage: 8 async b128 loads per thread (4 A + 4 B), ASYNCcnt-tracked.
  auto issue = [&](int buf, int kb) {
#pragma unroll
    for (int c = 0; c < 4; ++c) {
      __builtin_amdgcn_global_load_async_to_lds_b128(
          (AS1 i4v*)(S + (size_t)(mBase + srow[c]) * KDIM + kb + skp[c]),
          (AS3 i4v*)(&AsBuf[buf][srow[c] * LDAH + skp[c]]), 0, 0);
      __builtin_amdgcn_global_load_async_to_lds_b128(
          (AS1 i4v*)(W2 + (size_t)(nBase + srow[c]) * KDIM + kb + skp[c]),
          (AS3 i4v*)(&BsBuf[buf][srow[c] * LDAH + skp[c]]), 0, 0);
    }
  };

  v8f acc[2][4];
#pragma unroll
  for (int mi = 0; mi < 2; ++mi)
#pragma unroll
    for (int ni = 0; ni < 4; ++ni)
      acc[mi][ni] = (v8f){0.f, 0.f, 0.f, 0.f, 0.f, 0.f, 0.f, 0.f};

  // Prologue: two stages in flight.
  issue(0, 0);
  issue(1, BK);

  int buf = 0;
  for (int kb = 0; kb < KDIM; kb += BK) {
    // Own stage-kb loads complete (two stages outstanding in steady state).
    if (kb + BK < KDIM) asm volatile("s_wait_asynccnt 0x8" ::: "memory");
    else                asm volatile("s_wait_asynccnt 0x0" ::: "memory");
    // All waves' stage-kb loads complete, all waves done reading buf (kb-1).
    asm volatile("s_barrier_signal -1\n\ts_barrier_wait -1" ::: "memory");

    // Refill the buffer that was consumed at iteration kb-1.
    if (kb + 2 * BK < KDIM) {
      int nb = buf + 2; if (nb >= NSTG) nb -= NSTG;
      issue(nb, kb + 2 * BK);
    }

    const _Float16* As = AsBuf[buf];
    const _Float16* Bs = BsBuf[buf];

#pragma unroll
    for (int ks = 0; ks < BK; ks += 32) {
      // A fragments (16x32 f16): lane l = row l%16, K halves per ISA layout.
      F16x16 af[2];
#pragma unroll
      for (int mi = 0; mi < 2; ++mi) {
        const int r = wm * 32 + mi * 16 + lr;
        af[mi].h[0] = *(const v8h*)(&As[r * LDAH + ks + 8 * half]);
        af[mi].h[1] = *(const v8h*)(&As[r * LDAH + ks + 16 + 8 * half]);
      }
      // B fragments (32x16 f16): lane l = col l%16, K halves per ISA layout (same as A).
      F16x16 bf[4];
#pragma unroll
      for (int ni = 0; ni < 4; ++ni) {
        const int c = wn * 64 + ni * 16 + lr;
        bf[ni].h[0] = *(const v8h*)(&Bs[c * LDAH + ks + 8 * half]);
        bf[ni].h[1] = *(const v8h*)(&Bs[c * LDAH + ks + 16 + 8 * half]);
      }
#pragma unroll
      for (int mi = 0; mi < 2; ++mi)
#pragma unroll
        for (int ni = 0; ni < 4; ++ni)
          acc[mi][ni] = __builtin_amdgcn_wmma_f32_16x16x32_f16(
              false, af[mi].v, false, bf[ni].v, (short)0, acc[mi][ni], false, false);
    }

    if (++buf >= NSTG) buf = 0;
  }

  // Epilogue: C/D layout lane l, VGPR v -> row 8*(l/16)+v, col l%16.
#pragma unroll
  for (int mi = 0; mi < 2; ++mi) {
#pragma unroll
    for (int ni = 0; ni < 4; ++ni) {
      const int gRow0 = mBase + wm * 32 + mi * 16 + 8 * half;
      const int gCol  = nBase + wn * 64 + ni * 16 + lr;
#pragma unroll
      for (int v = 0; v < 8; ++v)
        out[(size_t)(gRow0 + v) * OUTSZ + gCol] = acc[mi][ni][v];
    }
  }
}

// ---------------------------------------------------------------------------
extern "C" void kernel_launch(void* const* d_in, const int* in_sizes, int n_in,
                              void* d_out, int out_size, void* d_ws, size_t ws_size,
                              hipStream_t stream) {
  const float* x    = (const float*)d_in[0];   // [8192, 512]
  const float* w    = (const float*)d_in[1];   // [512, 512]
  const float* coef = (const float*)d_in[2];   // [512, 512, 8]
  float* out        = (float*)d_out;           // [8192, 512]

  char* ws = (char*)d_ws;
  _Float16* Smat = (_Float16*)ws;                                             // 64 MB
  _Float16* W2   = (_Float16*)(ws + (size_t)BATCH * KDIM * sizeof(_Float16)); // 4 MB

  kan_fuse_w<<<(OUTSZ * INSZ) / 256, 256, 0, stream>>>(w, coef, W2);
  kan_basis<<<(BATCH * INSZ) / 256, 256, 0, stream>>>(x, Smat);
  dim3 grid(OUTSZ / BN, BATCH / BM);   // (4, 64)
  kan_wmma_gemm<<<grid, 256, 0, stream>>>(Smat, W2, out);
}
